// NTM_6442450944382
// MI455X (gfx1250) — compile-verified, matrix operand order checked by reading, not run on hardware
//
#include <hip/hip_runtime.h>
#include <hip/hip_bf16.h>

// ---------------------------------------------------------------------------
// NTM forward for MI455X (gfx1250, wave32, WMMA).
//
// 2 persistent workgroups x 16 batches (M=16 WMMA tiles), all 256 timesteps
// inside one kernel, bf16 WMMA (f32 accum) for the three per-step GEMMs,
// VALU f32 attention, state in LDS (196KB/WGP), weights pre-packed into
// per-lane WMMA B-fragment layout. Software-pipelined B/A fragment loads.
//
// Round-2 change: WGP-scope (scope 0) streaming prefetch via inline asm —
// __builtin_prefetch can only express SE/DEV scope, which never fills the
// WGP$; scope-0 global_prefetch_b8 pulls the weight stream into all cache
// levels ~8 k-steps ahead of the consuming loads.
// ---------------------------------------------------------------------------

typedef __bf16 v16bf __attribute__((ext_vector_type(16)));
typedef __bf16 v8bf  __attribute__((ext_vector_type(8)));
typedef float  v8f   __attribute__((ext_vector_type(8)));

#define DINL __device__ __forceinline__

constexpr int Bc = 32, Lc = 256, Ec = 256, HIDc = 512, MEMc = 128, HSc = 64, OUTc = 256;
constexpr int GB   = 16;    // batches per workgroup (== WMMA M tile)
constexpr int KCAT = 1024;  // concat K for gates GEMM: [x(256) | rc(256) | h(512)]
constexpr int NHEADP = 576; // 4 heads * 144 (129 used per head, padded)
constexpr int HSTR   = 144;

// workspace layout (bytes). Fragment = 32 lanes * 16 bf16 = 1024 B.
constexpr size_t WCAT_OFF  = 0;                          // 4096 frags (128 ntiles * 32 ksteps)
constexpr size_t WCAT_SZ   = (size_t)4096 * 1024;
constexpr size_t WHEAD_OFF = WCAT_OFF + WCAT_SZ;         // 576 frags (36 * 16)
constexpr size_t WHEAD_SZ  = (size_t)576 * 1024;
constexpr size_t WOUT_OFF  = WHEAD_OFF + WHEAD_SZ;       // 384 frags (16 * 24)
constexpr size_t WOUT_SZ   = (size_t)384 * 1024;
constexpr size_t BIASG_OFF = WOUT_OFF + WOUT_SZ;         // 2048 f32
constexpr size_t BHEAD_OFF = BIASG_OFF + 2048 * 4;       // 576 f32
// total ~5.19 MB of d_ws

// d_out layout (f32 elements): outputs | mem | h | c
constexpr size_t O_MEM = (size_t)Bc * Lc * OUTc;          // 2097152
constexpr size_t O_H   = O_MEM + (size_t)Bc * MEMc * HSc; // 2359296
constexpr size_t O_C   = O_H + (size_t)Bc * HIDc;         // 2375680

// k index inside a 16x16x32 fragment for (lane, elem): per-lane K set is
// {off..off+7} U {off+16..off+23}, off = (lane>=16 ? 8 : 0)   [ISA 7.12.2]
DINL int frag_k(int lane, int e) { return ((lane >> 4) * 8) + (e & 7) + ((e >> 3) * 16); }

// WGP-scope (scope 0) prefetch: fills WGP$ + L2; stops on WGP$ hit.
DINL void prefetch_wgp(const void* p) {
  asm volatile("global_prefetch_b8 %0, off" :: "v"(p));
}

// ------------------------- weight packing kernels --------------------------

__global__ void pack_wcat(const float* __restrict__ Wih, const float* __restrict__ Whh,
                          __bf16* __restrict__ dst) {
  int gid = blockIdx.x * blockDim.x + threadIdx.x;     // 4096*512 = 2097152
  int e = gid & 15, lane = (gid >> 4) & 31, frag = gid >> 9;
  int kk = frag & 31, nt = frag >> 5;                  // frag = ntile*32 + kstep
  int n = nt * 16 + (lane & 15);
  int k = kk * 32 + frag_k(lane, e);                   // 0..1023
  float v = (k < 512) ? Wih[n * 512 + k] : Whh[n * 512 + (k - 512)];
  dst[(size_t)frag * 512 + lane * 16 + e] = (__bf16)v;
}

__global__ void pack_whead(const float* __restrict__ Wh, __bf16* __restrict__ dst) {
  int gid = blockIdx.x * blockDim.x + threadIdx.x;     // 576*512 = 294912
  int e = gid & 15, lane = (gid >> 4) & 31, frag = gid >> 9;
  int kk = frag & 15, nt = frag >> 4;                  // frag = ntile*16 + kstep
  int np = nt * 16 + (lane & 15);                      // padded head output index
  int head = np / HSTR, r = np % HSTR;                 // r<129 used (64 rk, 64 wk, 1 strength)
  int k = kk * 32 + frag_k(lane, e);                   // 0..511
  float v = (r < 129) ? Wh[(head * 195 + r) * 512 + k] : 0.f;
  dst[(size_t)frag * 512 + lane * 16 + e] = (__bf16)v;
}

__global__ void pack_wout(const float* __restrict__ Wo, __bf16* __restrict__ dst) {
  int gid = blockIdx.x * blockDim.x + threadIdx.x;     // 384*512 = 196608
  int e = gid & 15, lane = (gid >> 4) & 31, frag = gid >> 9;
  int kk = frag % 24, nt = frag / 24;                  // frag = ntile*24 + kstep
  int n = nt * 16 + (lane & 15);
  int Kp = kk * 32 + frag_k(lane, e);                  // 0..767 over Abuf[256..1024)
  int col = (Kp < 256) ? (512 + Kp) : (Kp - 256);      // rc -> W_out[:,512+], h -> W_out[:,0..512)
  dst[(size_t)frag * 512 + lane * 16 + e] = (__bf16)Wo[n * 768 + col];
}

__global__ void pack_bias(const float* __restrict__ bih, const float* __restrict__ bhh,
                          const float* __restrict__ bhead,
                          float* __restrict__ biasG, float* __restrict__ bheadP) {
  int gid = blockIdx.x * blockDim.x + threadIdx.x;
  if (gid < 2048) {
    biasG[gid] = bih[gid] + bhh[gid];
  } else if (gid < 2048 + NHEADP) {
    int np = gid - 2048;
    int head = np / HSTR, r = np % HSTR;
    bheadP[np] = (r < 129) ? bhead[head * 195 + r] : 0.f;
  }
}

// ----------------------------- main kernel ---------------------------------

DINL float sigm(float x) { return 1.0f / (1.0f + __expf(-x)); }

// A-fragment gather from row-major bf16 LDS activations (ISA 7.12.2 layout)
DINL v16bf loadA(const __bf16* arow, int k0) {
  v8bf lo = *(const v8bf*)(arow + k0);
  v8bf hi = *(const v8bf*)(arow + k0 + 16);
  return __builtin_shufflevector(lo, hi, 0, 1, 2, 3, 4, 5, 6, 7, 8, 9, 10, 11, 12, 13, 14, 15);
}

DINL v16bf loadB(const __bf16* p) { return *(const v16bf*)p; }

__global__ __launch_bounds__(512, 1) void ntm_persistent(
    const float* __restrict__ x,
    const __bf16* __restrict__ wcat, const __bf16* __restrict__ whead,
    const __bf16* __restrict__ wout,
    const float* __restrict__ biasG, const float* __restrict__ bheadP,
    const float* __restrict__ bout, float* __restrict__ out) {
  extern __shared__ char lds[];
  __bf16* Abuf = (__bf16*)lds;             // [16][1024] bf16: x|rc|h   32 KB
  float*  cS   = (float*)(lds + 32768);    // [16][512]                 32 KB
  float*  hS   = (float*)(lds + 65536);    // [16][512]                 32 KB
  float*  hpS  = (float*)(lds + 98304);    // [16][576] head outputs    36 KB
  float*  wS   = (float*)(lds + 135168);   // [16][8][128] rw|ww        64 KB

  const int tid  = threadIdx.x;
  const int wave = tid >> 5;
  const int lane = tid & 31;
  const int b0   = blockIdx.x * GB;
  float* memG = out + O_MEM + (size_t)b0 * MEMc * HSc;

  // --- init state (mem0 = h0 = c0 = rc0 = 0); deterministic per launch ---
  for (int i = tid; i < GB * KCAT; i += 512) Abuf[i] = (__bf16)0.f;
  for (int i = tid; i < GB * HIDc; i += 512) { cS[i] = 0.f; hS[i] = 0.f; }
  for (int i = tid; i < GB * MEMc * HSc; i += 512) memG[i] = 0.f;
  __syncthreads();

  const int row = lane & 15;
  const int off = (lane >> 4) * 8;
  const __bf16* arow = Abuf + row * KCAT;

  for (int t = 0; t < Lc; t++) {
    // ---- phase 1: x_t -> Abuf[:,0:256] (bf16) ----
    for (int i = tid; i < GB * Ec; i += 512) {
      int bb = i >> 8, e = i & 255;
      Abuf[bb * KCAT + e] = (__bf16)x[((size_t)(b0 + bb) * Lc + t) * Ec + e];
    }
    __syncthreads();

    // ---- phase 2: gates = [x|rc|h] @ [W_ih|W_hh]^T, fused LSTM update ----
    // Software-pipelined: B fragments (4 gates) and A fragment for k-step
    // kk+1 are loaded into fresh registers before the 4 WMMAs of k-step kk,
    // so each WMMA waits only on loads issued one full k-step earlier.
    for (int rep = 0; rep < 2; rep++) {
      int jt = wave + rep * 16;  // hid tile 0..31; gate g at ntile jt + g*32
      v8f ai = {}, af = {}, ag = {}, ao = {};
      const __bf16* bp0 = wcat + ((size_t)(jt)*32) * 512 + lane * 16;
      const __bf16* bp1 = wcat + ((size_t)(jt + 32) * 32) * 512 + lane * 16;
      const __bf16* bp2 = wcat + ((size_t)(jt + 64) * 32) * 512 + lane * 16;
      const __bf16* bp3 = wcat + ((size_t)(jt + 96) * 32) * 512 + lane * 16;
      v16bf b0c = loadB(bp0), b1c = loadB(bp1), b2c = loadB(bp2), b3c = loadB(bp3);
      v16bf avc = loadA(arow, off);
      for (int kk = 0; kk < 32; kk++) {
        // WGP-scope prefetch 8 k-steps ahead on all four gate streams
        prefetch_wgp(bp0 + (kk + 8) * 512);
        prefetch_wgp(bp1 + (kk + 8) * 512);
        prefetch_wgp(bp2 + (kk + 8) * 512);
        prefetch_wgp(bp3 + (kk + 8) * 512);
        // issue next k-step loads (one-past-end lands in adjacent ws/LDS: harmless)
        v16bf b0n = loadB(bp0 + (kk + 1) * 512);
        v16bf b1n = loadB(bp1 + (kk + 1) * 512);
        v16bf b2n = loadB(bp2 + (kk + 1) * 512);
        v16bf b3n = loadB(bp3 + (kk + 1) * 512);
        v16bf avn = loadA(arow, (kk + 1) * 32 + off);
        ai = __builtin_amdgcn_wmma_f32_16x16x32_bf16(false, avc, false, b0c, (short)0, ai, false, false);
        af = __builtin_amdgcn_wmma_f32_16x16x32_bf16(false, avc, false, b1c, (short)0, af, false, false);
        ag = __builtin_amdgcn_wmma_f32_16x16x32_bf16(false, avc, false, b2c, (short)0, ag, false, false);
        ao = __builtin_amdgcn_wmma_f32_16x16x32_bf16(false, avc, false, b3c, (short)0, ao, false, false);
        b0c = b0n; b1c = b1n; b2c = b2n; b3c = b3n; avc = avn;
      }
      int j = jt * 16 + (lane & 15);
      float bi = biasG[j], bf = biasG[512 + j], bg = biasG[1024 + j], bo = biasG[1536 + j];
#pragma unroll
      for (int r = 0; r < 8; r++) {
        int brow = r + ((lane >> 4) << 3);  // C/D layout: M = r + 8*(lane/16)
        float iv = sigm(ai[r] + bi);
        float fv = sigm(af[r] + bf);
        float gv = tanhf(ag[r] + bg);
        float ov = sigm(ao[r] + bo);
        float cn = fv * cS[brow * HIDc + j] + iv * gv;
        cS[brow * HIDc + j] = cn;
        float hn = ov * tanhf(cn);
        hS[brow * HIDc + j] = hn;
        Abuf[brow * KCAT + 512 + j] = (__bf16)hn;
      }
    }
    __syncthreads();

    // ---- phase 3: head GEMM hp = h @ W_head^T (gathered/padded N=576) ----
    for (int nt = wave; nt < 36; nt += 16) {
      v8f acc = {};
      const __bf16* bp = whead + ((size_t)nt * 16) * 512 + lane * 16;
      v16bf bc = loadB(bp);
      v16bf avc = loadA(arow, 512 + off);
      for (int kk = 0; kk < 16; kk++) {
        prefetch_wgp(bp + (kk + 8) * 512);
        v16bf bn = loadB(bp + (kk + 1) * 512);           // frag 576 -> wout region, harmless
        v16bf avn = loadA(arow, 512 + (kk + 1) * 32 + off);
        acc = __builtin_amdgcn_wmma_f32_16x16x32_bf16(false, avc, false, bc, (short)0, acc, false, false);
        bc = bn; avc = avn;
      }
      int np = nt * 16 + (lane & 15);
      float bh = bheadP[np];
#pragma unroll
      for (int r = 0; r < 8; r++) {
        int brow = r + ((lane >> 4) << 3);
        hpS[brow * NHEADP + np] = acc[r] + bh;
      }
    }
    __syncthreads();

    // ---- phase 4: attention; one wave per batch element ----
    {
      int bb = wave;
      const float* hpb = hpS + bb * NHEADP;
      float* memB = memG + (size_t)bb * MEMc * HSc;
      // pass 1: similarities (8 keys: 4 read + 4 write) over 128 mem rows
      float s[8][4];
#pragma unroll
      for (int q = 0; q < 4; q++) {
        int m = lane * 4 + q;
        const float4* pm = (const float4*)(memB + m * HSc);
        float acc[8] = {0, 0, 0, 0, 0, 0, 0, 0};
        for (int h4 = 0; h4 < 16; h4++) {
          float4 mv = pm[h4];
#pragma unroll
          for (int n = 0; n < 8; n++) {
            int koff = (n < 4) ? n * HSTR : (n - 4) * HSTR + 64;
            float4 kv = *(const float4*)(hpb + koff + h4 * 4);
            acc[n] += kv.x * mv.x + kv.y * mv.y + kv.z * mv.z + kv.w * mv.w;
          }
        }
#pragma unroll
        for (int n = 0; n < 8; n++) s[n][q] = acc[n] * 0.125f;  // 1/sqrt(64)
      }
      // softmax over m=128 per key (wave32 shuffle reductions)
#pragma unroll
      for (int n = 0; n < 8; n++) {
        float mx = fmaxf(fmaxf(s[n][0], s[n][1]), fmaxf(s[n][2], s[n][3]));
        for (int d = 16; d >= 1; d >>= 1) mx = fmaxf(mx, __shfl_xor(mx, d, 32));
        float sum = 0.f;
#pragma unroll
        for (int q = 0; q < 4; q++) { s[n][q] = __expf(s[n][q] - mx); sum += s[n][q]; }
        for (int d = 16; d >= 1; d >>= 1) sum += __shfl_xor(sum, d, 32);
        float inv = 1.0f / sum;
        *(float4*)(wS + bb * 1024 + n * 128 + lane * 4) =
            make_float4(s[n][0] * inv, s[n][1] * inv, s[n][2] * inv, s[n][3] * inv);
      }
      // pass 2: rc = rw@mem (old mem), mem += ww^T @ (ws * wkeys); lane owns 2 h cols
      int h0 = lane * 2;
      float wsk0[4], wsk1[4];
#pragma unroll
      for (int n = 0; n < 4; n++) {
        float wsn = sigm(hpb[n * HSTR + 128]);
        wsk0[n] = wsn * hpb[n * HSTR + 64 + h0];
        wsk1[n] = wsn * hpb[n * HSTR + 64 + h0 + 1];
      }
      float rc0[4] = {0, 0, 0, 0}, rc1[4] = {0, 0, 0, 0};
      const float* wSb = wS + bb * 1024;
      for (int m = 0; m < MEMc; m++) {
        float2 mv = *(float2*)(memB + m * HSc + h0);
        float d0 = 0.f, d1 = 0.f;
#pragma unroll
        for (int n = 0; n < 4; n++) {
          float rw = wSb[n * 128 + m];
          rc0[n] += rw * mv.x; rc1[n] += rw * mv.y;
          float ww = wSb[(4 + n) * 128 + m];
          d0 += ww * wsk0[n]; d1 += ww * wsk1[n];
        }
        *(float2*)(memB + m * HSc + h0) = make_float2(mv.x + d0, mv.y + d1);
      }
      __bf16* rcd = Abuf + bb * KCAT + 256;
#pragma unroll
      for (int n = 0; n < 4; n++) {
        rcd[n * HSc + h0]     = (__bf16)rc0[n];
        rcd[n * HSc + h0 + 1] = (__bf16)rc1[n];
      }
    }
    __syncthreads();

    // ---- phase 5: out = [h|rc] @ W_out^T + b_out (N=256, 16 tiles/16 waves) ----
    {
      int nt = wave;
      v8f acc = {};
      const __bf16* bp = wout + ((size_t)nt * 24) * 512 + lane * 16;
      v16bf bc = loadB(bp);
      v16bf avc = loadA(arow, 256 + off);
      for (int kk = 0; kk < 24; kk++) {
        prefetch_wgp(bp + (kk + 8) * 512);
        v16bf bn = loadB(bp + (kk + 1) * 512);           // frag 384 -> bias region, harmless
        v16bf avn = loadA(arow, 256 + (kk + 1) * 32 + off);
        acc = __builtin_amdgcn_wmma_f32_16x16x32_bf16(false, avc, false, bc, (short)0, acc, false, false);
        bc = bn; avc = avn;
      }
      float bo = bout[nt * 16 + (lane & 15)];
#pragma unroll
      for (int r = 0; r < 8; r++) {
        int brow = r + ((lane >> 4) << 3);
        out[((size_t)(b0 + brow) * Lc + t) * OUTc + nt * 16 + (lane & 15)] = acc[r] + bo;
      }
    }
    __syncthreads();
  }

  // ---- finalize h, c outputs ----
  for (int i = tid; i < GB * HIDc; i += 512) {
    int bb = i >> 9, j = i & 511;
    out[O_H + (size_t)(b0 + bb) * HIDc + j] = hS[i];
    out[O_C + (size_t)(b0 + bb) * HIDc + j] = cS[i];
  }
}

// ------------------------------- launcher ----------------------------------

extern "C" void kernel_launch(void* const* d_in, const int* in_sizes, int n_in,
                              void* d_out, int out_size, void* d_ws, size_t ws_size,
                              hipStream_t stream) {
  (void)in_sizes; (void)n_in; (void)out_size; (void)ws_size;
  const float* x     = (const float*)d_in[0];
  const float* Wih   = (const float*)d_in[1];
  const float* Whh   = (const float*)d_in[2];
  const float* bih   = (const float*)d_in[3];
  const float* bhh   = (const float*)d_in[4];
  const float* Whead = (const float*)d_in[5];
  const float* bhead = (const float*)d_in[6];
  const float* Wout  = (const float*)d_in[7];
  const float* bout  = (const float*)d_in[8];
  float* out = (float*)d_out;
  char*  ws  = (char*)d_ws;  // needs ~5.2 MB

  __bf16* wcatP  = (__bf16*)(ws + WCAT_OFF);
  __bf16* wheadP = (__bf16*)(ws + WHEAD_OFF);
  __bf16* woutP  = (__bf16*)(ws + WOUT_OFF);
  float*  biasG  = (float*)(ws + BIASG_OFF);
  float*  bheadB = (float*)(ws + BHEAD_OFF);

  pack_wcat <<<8192, 256, 0, stream>>>(Wih, Whh, wcatP);
  pack_whead<<<1152, 256, 0, stream>>>(Whead, wheadP);
  pack_wout <<< 768, 256, 0, stream>>>(Wout, woutP);
  pack_bias <<<  11, 256, 0, stream>>>(bih, bhh, bhead, biasG, bheadB);

  constexpr size_t LDS_BYTES = 200704;  // 196 KB of the 320 KB/WGP LDS
  ntm_persistent<<<2, 512, LDS_BYTES, stream>>>(x, wcatP, wheadP, woutP,
                                                biasG, bheadB, bout, out);
}
